// GroupQueryAttention_86672440033875
// MI455X (gfx1250) — matrix operand order repaired and checked by reading, not compiled, not run on hardware
//
#include <hip/hip_runtime.h>
#include <hip/hip_bf16.h>
#include <cstdint>
#include <cstddef>

typedef __bf16 bf16;
typedef __attribute__((ext_vector_type(16))) __bf16 v16bf;
typedef __attribute__((ext_vector_type(8)))  __bf16 v8bf;
typedef __attribute__((ext_vector_type(8)))  float  v8f;
typedef __attribute__((ext_vector_type(4)))  int    v4i;

#define D_MODEL 2048
#define HEAD    64
#define NH      32
#define NKV     8
#define GRP     4
#define BSZ     2
#define TSZ     2048
#define MROWS   (BSZ*TSZ)                 // 4096
#define NQKV    (NH*HEAD + 2*NKV*HEAD)    // 3072

// ---------------------------------------------------------------------------
// CDNA5 async global->LDS copy (ASYNCcnt path), with safe fallback
// ---------------------------------------------------------------------------
#if defined(__has_builtin)
#if __has_builtin(__builtin_amdgcn_global_load_async_to_lds_b128)
#define HAVE_ASYNC_LDS 1
#endif
#endif
#ifndef HAVE_ASYNC_LDS
#define HAVE_ASYNC_LDS 0
#endif

typedef __attribute__((address_space(1))) v4i gv4i;   // global int4
typedef __attribute__((address_space(3))) v4i lv4i;   // LDS int4

__device__ __forceinline__ void async_copy16(const bf16* g, bf16* l) {
#if HAVE_ASYNC_LDS
    __builtin_amdgcn_global_load_async_to_lds_b128((gv4i*)g, (lv4i*)l, 0, 0);
#else
    *(v8bf*)l = *(const v8bf*)g;
#endif
}

__device__ __forceinline__ void async_wait0() {
#if HAVE_ASYNC_LDS
#if __has_builtin(__builtin_amdgcn_s_wait_asynccnt)
    __builtin_amdgcn_s_wait_asynccnt(0);
#else
    asm volatile("s_wait_asynccnt 0" ::: "memory");
#endif
#endif
}

// ---------------------------------------------------------------------------
// fp32 -> bf16 conversions
// ---------------------------------------------------------------------------
__global__ void cvt_x(const float* __restrict__ x, bf16* __restrict__ xb) {
    size_t i = (size_t)blockIdx.x * blockDim.x + threadIdx.x;  // MROWS*D_MODEL
    xb[i] = (bf16)x[i];
}

__global__ void cvt_w(const float* __restrict__ Wq, const float* __restrict__ Wk,
                      const float* __restrict__ Wv, bf16* __restrict__ wb) {
    size_t i = (size_t)blockIdx.x * blockDim.x + threadIdx.x;  // NQKV*D_MODEL
    int r = (int)(i / D_MODEL);
    int c = (int)(i % D_MODEL);
    float v;
    if (r < NH*HEAD)                 v = Wq[(size_t)r * D_MODEL + c];
    else if (r < NH*HEAD + NKV*HEAD) v = Wk[(size_t)(r - NH*HEAD) * D_MODEL + c];
    else                             v = Wv[(size_t)(r - NH*HEAD - NKV*HEAD) * D_MODEL + c];
    wb[i] = (bf16)v;
}

// ---------------------------------------------------------------------------
// QKV projection GEMM: C[4096][3072] = Xbf[4096][2048] * Wbf[3072][2048]^T
// Block tile 128(M) x 64(N), 8 waves, wave tile 32x32 (2x2 WMMA frags), K=32.
// Double-buffered LDS fed by async global->LDS copies: 1 barrier / K-step.
// ---------------------------------------------------------------------------
__global__ __launch_bounds__(256) void qkv_gemm(const bf16* __restrict__ X,
                                                const bf16* __restrict__ W,
                                                float* __restrict__ C) {
    __shared__ __align__(16) bf16 lx[2][128][40];   // +8 pad vs bank conflicts
    __shared__ __align__(16) bf16 lw[2][64][40];
    const int tid  = threadIdx.x;
    const int wid  = tid >> 5;
    const int lane = tid & 31;
    const int l16  = lane & 15;
    const int lh   = lane >> 4;          // which 16-lane half
    const int bm = blockIdx.x % (MROWS / 128);
    const int bn = blockIdx.x / (MROWS / 128);
    const int m0 = bm * 128, n0 = bn * 64;
    const int wm = (wid & 3) * 32;       // wave M offset: 0..96
    const int wn = (wid >> 2) * 32;      // wave N offset: 0 or 32

    // per-thread cooperative-copy coordinates
    const int rX = tid >> 1, segX = (tid & 1) * 16;   // X: 128 rows x 32 cols
    const int rW = tid >> 2, segW = (tid & 3) * 8;    // W:  64 rows x 32 cols
    const bf16* gX = X + (size_t)(m0 + rX) * D_MODEL + segX;
    const bf16* gW = W + (size_t)(n0 + rW) * D_MODEL + segW;

    v8f acc[2][2] = {};

    // prologue: fill buffer 0
    async_copy16(gX,     &lx[0][rX][segX]);
    async_copy16(gX + 8, &lx[0][rX][segX + 8]);
    async_copy16(gW,     &lw[0][rW][segW]);
    async_wait0();
    __syncthreads();

    int cur = 0;
    for (int k0 = 0; k0 < D_MODEL; k0 += 32) {
        // kick off next tile into the other buffer (overlaps with WMMA below)
        if (k0 + 32 < D_MODEL) {
            int nxt = cur ^ 1, kn = k0 + 32;
            async_copy16(gX + kn,     &lx[nxt][rX][segX]);
            async_copy16(gX + kn + 8, &lx[nxt][rX][segX + 8]);
            async_copy16(gW + kn,     &lw[nxt][rW][segW]);
        }
        // gather WMMA fragments from LDS (ISA 7.12.2 layouts)
        v16bf a[2], b[2];
#pragma unroll
        for (int f = 0; f < 2; ++f) {
            int row = wm + f * 16 + l16;
            union { v16bf v; v8bf h[2]; } ua;
            ua.h[0] = *(const v8bf*)&lx[cur][row][lh * 8];
            ua.h[1] = *(const v8bf*)&lx[cur][row][16 + lh * 8];
            a[f] = ua.v;
            int col = wn + f * 16 + l16;
            union { v16bf v; v8bf h[2]; } ub;
            ub.h[0] = *(const v8bf*)&lw[cur][col][lh * 8];
            ub.h[1] = *(const v8bf*)&lw[cur][col][16 + lh * 8];
            b[f] = ub.v;
        }
#pragma unroll
        for (int fm = 0; fm < 2; ++fm)
#pragma unroll
            for (int fn = 0; fn < 2; ++fn)
                acc[fm][fn] = __builtin_amdgcn_wmma_f32_16x16x32_bf16(
                    false, a[fm], false, b[fn], (short)0, acc[fm][fn], false, false);
        // next buffer ready + everyone done reading current buffer
        async_wait0();
        __syncthreads();
        cur ^= 1;
    }
#pragma unroll
    for (int fm = 0; fm < 2; ++fm)
#pragma unroll
        for (int fn = 0; fn < 2; ++fn)
#pragma unroll
            for (int r = 0; r < 8; ++r) {
                int row = m0 + wm + fm * 16 + lh * 8 + r;
                int col = n0 + wn + fn * 16 + l16;
                C[(size_t)row * NQKV + col] = acc[fm][fn][r];
            }
}

// ---------------------------------------------------------------------------
// RoPE + layout shuffles (Q gets the 1/sqrt(64) scale folded in)
// ---------------------------------------------------------------------------
__global__ void rope_q(const float* __restrict__ qkv, const float* __restrict__ cs,
                       const float* __restrict__ sn, bf16* __restrict__ Qo) {
    int tid = blockIdx.x * blockDim.x + threadIdx.x;   // BSZ*TSZ*NH*32 = 2^22
    int i = tid & 31;
    int h = (tid >> 5) & 31;
    int t = (tid >> 10) & (TSZ - 1);
    int b = tid >> 21;
    size_t row  = (size_t)(b * TSZ + t);
    size_t base = row * NQKV + h * HEAD + 2 * i;
    float x1 = qkv[base], x2 = qkv[base + 1];
    float c = cs[t * 32 + i], s = sn[t * 32 + i];
    size_t ob = ((size_t)((b * NH + h) * TSZ + t)) * HEAD + 2 * i;
    Qo[ob]     = (bf16)((x1 * c - x2 * s) * 0.125f);
    Qo[ob + 1] = (bf16)((x1 * s + x2 * c) * 0.125f);
}

__global__ void rope_k(const float* __restrict__ qkv, const float* __restrict__ cs,
                       const float* __restrict__ sn, bf16* __restrict__ Ko) {
    int tid = blockIdx.x * blockDim.x + threadIdx.x;   // BSZ*TSZ*NKV*32 = 2^20
    int i = tid & 31;
    int h = (tid >> 5) & 7;
    int t = (tid >> 8) & (TSZ - 1);
    int b = tid >> 19;
    size_t row  = (size_t)(b * TSZ + t);
    size_t base = row * NQKV + NH * HEAD + h * HEAD + 2 * i;
    float x1 = qkv[base], x2 = qkv[base + 1];
    float c = cs[t * 32 + i], s = sn[t * 32 + i];
    size_t ob = ((size_t)((b * NKV + h) * TSZ + t)) * HEAD + 2 * i;
    Ko[ob]     = (bf16)(x1 * c - x2 * s);
    Ko[ob + 1] = (bf16)(x1 * s + x2 * c);
}

__global__ void cvt_v(const float* __restrict__ qkv, bf16* __restrict__ Vt) {
    int tid = blockIdx.x * blockDim.x + threadIdx.x;   // BSZ*TSZ*NKV*HEAD = 2^21
    int d = tid & 63;
    int h = (tid >> 6) & 7;
    int t = (tid >> 9) & (TSZ - 1);
    int b = tid >> 20;
    size_t row = (size_t)(b * TSZ + t);
    float v = qkv[row * NQKV + (NH + NKV) * HEAD + h * HEAD + d];
    // store V transposed: [b][kvh][d][t] so PV B-fragments are contiguous
    Vt[((size_t)((b * NKV + h) * HEAD + d)) * TSZ + t] = (bf16)v;
}

// ---------------------------------------------------------------------------
// Causal flash attention: one wave per 16-query tile, 32-key steps.
// ---------------------------------------------------------------------------
__global__ __launch_bounds__(128) void attn(const bf16* __restrict__ Q,
                                            const bf16* __restrict__ Kc,
                                            const bf16* __restrict__ Vt,
                                            float* __restrict__ O) {
    __shared__ __align__(16) bf16 pbuf[4][16 * 32];   // per-wave P staging
    const int wid  = threadIdx.x >> 5;
    const int lane = threadIdx.x & 31;
    const int l16  = lane & 15, lh = lane >> 4;
    int gid = blockIdx.x * 4 + wid;                    // 8192 wave tiles
    int b   = gid / (NH * (TSZ / 16));
    int rem = gid % (NH * (TSZ / 16));
    int h   = rem / (TSZ / 16);
    int q0  = (rem % (TSZ / 16)) * 16;
    int kvh = h / GRP;
    const bf16* qp = Q  + ((size_t)(b * NH + h) * TSZ) * HEAD;
    const bf16* kp = Kc + ((size_t)(b * NKV + kvh) * TSZ) * HEAD;
    const bf16* vp = Vt + ((size_t)(b * NKV + kvh) * HEAD) * TSZ;

    // Q A-fragments (d split 0..31 / 32..63), pre-scaled by 0.125
    v16bf qa[2];
#pragma unroll
    for (int c = 0; c < 2; ++c) {
        union { v16bf v; v8bf hh[2]; } u;
        const bf16* p = qp + (size_t)(q0 + l16) * HEAD + c * 32;
        u.hh[0] = *(const v8bf*)(p + lh * 8);
        u.hh[1] = *(const v8bf*)(p + 16 + lh * 8);
        qa[c] = u.v;
    }

    float rm[8], rl[8];
#pragma unroll
    for (int r = 0; r < 8; ++r) { rm[r] = -1e30f; rl[r] = 0.f; }
    v8f o[4] = {};

    int nkb = ((q0 + 15) >> 5) + 1;
    for (int kb = 0; kb < nkb; ++kb) {
        int n0 = kb * 32;
        // S = Q K^T for 2 x 16 keys
        v8f s01[2];
#pragma unroll
        for (int g = 0; g < 2; ++g) {
            v8f s = {};
#pragma unroll
            for (int c = 0; c < 2; ++c) {
                union { v16bf v; v8bf hh[2]; } u;
                const bf16* p = kp + (size_t)(n0 + g * 16 + l16) * HEAD + c * 32 + lh * 8;
                u.hh[0] = *(const v8bf*)p;
                u.hh[1] = *(const v8bf*)(p + 16);
                s = __builtin_amdgcn_wmma_f32_16x16x32_bf16(
                        false, qa[c], false, u.v, (short)0, s, false, false);
            }
            s01[g] = s;
        }
        // online softmax (rows live per 16-lane half in C layout)
        int key0 = n0 + l16, key1 = key0 + 16;
        float p0[8], p1[8], corr[8];
#pragma unroll
        for (int r = 0; r < 8; ++r) {
            int qi = q0 + lh * 8 + r;
            float v0 = (key0 <= qi) ? s01[0][r] : -1e30f;
            float v1 = (key1 <= qi) ? s01[1][r] : -1e30f;
            float mx = fmaxf(v0, v1);
            mx = fmaxf(mx, __shfl_xor(mx, 1, 32));
            mx = fmaxf(mx, __shfl_xor(mx, 2, 32));
            mx = fmaxf(mx, __shfl_xor(mx, 4, 32));
            mx = fmaxf(mx, __shfl_xor(mx, 8, 32));
            float mn = fmaxf(rm[r], mx);
            corr[r] = __expf(rm[r] - mn);
            p0[r] = (key0 <= qi) ? __expf(v0 - mn) : 0.f;
            p1[r] = (key1 <= qi) ? __expf(v1 - mn) : 0.f;
            float sum = p0[r] + p1[r];
            sum += __shfl_xor(sum, 1, 32);
            sum += __shfl_xor(sum, 2, 32);
            sum += __shfl_xor(sum, 4, 32);
            sum += __shfl_xor(sum, 8, 32);
            rl[r] = rl[r] * corr[r] + sum;
            rm[r] = mn;
        }
#pragma unroll
        for (int f = 0; f < 4; ++f)
#pragma unroll
            for (int r = 0; r < 8; ++r)
                o[f][r] *= corr[r];
        // P: C layout -> A layout via wave-private LDS (DS is in-order per wave)
#pragma unroll
        for (int r = 0; r < 8; ++r) {
            pbuf[wid][(lh * 8 + r) * 32 + l16]      = (bf16)p0[r];
            pbuf[wid][(lh * 8 + r) * 32 + 16 + l16] = (bf16)p1[r];
        }
        __builtin_amdgcn_wave_barrier();
        asm volatile("s_wait_dscnt 0" ::: "memory");
        union { v16bf v; v8bf hh[2]; } pu;
        pu.hh[0] = *(const v8bf*)&pbuf[wid][l16 * 32 + lh * 8];
        pu.hh[1] = *(const v8bf*)&pbuf[wid][l16 * 32 + 16 + lh * 8];
        __builtin_amdgcn_wave_barrier();
        // O += P V  (V transposed: contiguous B-fragment loads)
#pragma unroll
        for (int f = 0; f < 4; ++f) {
            union { v16bf v; v8bf hh[2]; } vb;
            const bf16* p = vp + (size_t)(f * 16 + l16) * TSZ + n0 + lh * 8;
            vb.hh[0] = *(const v8bf*)p;
            vb.hh[1] = *(const v8bf*)(p + 16);
            o[f] = __builtin_amdgcn_wmma_f32_16x16x32_bf16(
                       false, pu.v, false, vb.v, (short)0, o[f], false, false);
        }
    }
    // epilogue: out[b][t][h*64+d] = O / rowsum
#pragma unroll
    for (int f = 0; f < 4; ++f)
#pragma unroll
        for (int r = 0; r < 8; ++r) {
            int t   = q0 + lh * 8 + r;
            int col = h * HEAD + f * 16 + l16;
            O[((size_t)(b * TSZ + t)) * (NH * HEAD) + col] = o[f][r] / rl[r];
        }
}

// ---------------------------------------------------------------------------
extern "C" void kernel_launch(void* const* d_in, const int* in_sizes, int n_in,
                              void* d_out, int out_size, void* d_ws, size_t ws_size,
                              hipStream_t stream) {
    const float* x  = (const float*)d_in[0];
    const float* Wq = (const float*)d_in[1];
    const float* Wk = (const float*)d_in[2];
    const float* Wv = (const float*)d_in[3];
    const float* cs = (const float*)d_in[4];
    const float* sn = (const float*)d_in[5];
    float* out = (float*)d_out;

    char* ws = (char*)d_ws;
    // workspace layout (total 104,857,600 B)
    bf16*  xb  = (bf16*)(ws);                          // 16,777,216
    bf16*  wb  = (bf16*)(ws + 16777216);               // 12,582,912
    float* qkv = (float*)(ws + 29360128);              // 50,331,648
    bf16*  qb  = (bf16*)(ws + 79691776);               // 16,777,216
    bf16*  kbuf= (bf16*)(ws + 96468992);               //  4,194,304
    bf16*  vt  = (bf16*)(ws + 100663296);              //  4,194,304

    cvt_x   <<<(MROWS * D_MODEL) / 256, 256, 0, stream>>>(x, xb);
    cvt_w   <<<(NQKV * D_MODEL) / 256, 256, 0, stream>>>(Wq, Wk, Wv, wb);
    qkv_gemm<<<(MROWS / 128) * (NQKV / 64), 256, 0, stream>>>(xb, wb, qkv);
    rope_q  <<<(BSZ * TSZ * NH * 32) / 256, 256, 0, stream>>>(qkv, cs, sn, qb);
    rope_k  <<<(BSZ * TSZ * NKV * 32) / 256, 256, 0, stream>>>(qkv, cs, sn, kbuf);
    cvt_v   <<<(BSZ * TSZ * NKV * HEAD) / 256, 256, 0, stream>>>(qkv, vt);
    attn    <<<(BSZ * NH * (TSZ / 16)) / 4, 128, 0, stream>>>(qb, kbuf, vt, out);
}
